// LTM_60292750901718
// MI455X (gfx1250) — matrix operand order repaired and checked by reading, not compiled, not run on hardware
//
#include <hip/hip_runtime.h>
#include <hip/hip_bf16.h>
#include <cstdint>

// ---------------------------------------------------------------------------
// Fused 4-gate LSTM-cell projection for MI455X (gfx1250, wave32, WMMA).
// Phase 1: one-shot fp32 -> bf16 conversion of x and W_{i,f,g,o} into d_ws.
// Phase 2: tiled GEMM, async global->LDS double-buffered staging, bf16 WMMA
//          with f32 accumulation, fused sigmoid/tanh LSTM epilogue.
// Tile: BM=256 x BN=32(x4 gates) x BK=64; 8 waves as 4(M) x 2(N); each wave
// holds 4 M-tiles so every B fragment feeds 4 WMMAs (ds:wmma = 1:1).
// ---------------------------------------------------------------------------

typedef __attribute__((ext_vector_type(8)))  float  v8f;
typedef __attribute__((ext_vector_type(8)))  __bf16 v8bf;
typedef __attribute__((ext_vector_type(16))) __bf16 v16bf;
typedef __attribute__((ext_vector_type(4)))  int    v4i;

// address-space-qualified pointer types for the async-to-LDS builtin
typedef __attribute__((address_space(1))) v4i glob_v4i;
typedef __attribute__((address_space(3))) v4i lds_v4i;

#define B_M 4096   // batch (GEMM M)
#define N_H 2048   // hidden (GEMM N, per gate)
#define K_D 2048   // input size (GEMM K)

#define BM 256     // block tile M
#define BN 32      // block tile N (per gate)
#define BK 64      // block tile K
#define LDK (BK + 8)   // padded LDS row stride in elements -> 144B (9x16B)

union Frag16 {
  v16bf v;
  struct { v8bf lo; v8bf hi; } h;
};

// fp32 -> packed 2x bf16 (round-to-nearest-even)
static __device__ __forceinline__ unsigned f2bf2(float a, float b) {
  unsigned ua = __builtin_bit_cast(unsigned, a);
  unsigned ub = __builtin_bit_cast(unsigned, b);
  ua = (ua + 0x7FFFu + ((ua >> 16) & 1u)) >> 16;
  ub = (ub + 0x7FFFu + ((ub >> 16) & 1u)) >> 16;
  return (ua & 0xFFFFu) | (ub << 16);
}

static __device__ __forceinline__ float sigmoid_f(float x) {
  return 1.0f / (1.0f + __expf(-x));
}
static __device__ __forceinline__ float tanh_f(float x) {
  float e = __expf(2.0f * x);
  return (e - 1.0f) / (e + 1.0f);
}

// ---- async global->LDS copy (16B per lane), with sync fallback ----
#if __has_builtin(__builtin_amdgcn_global_load_async_to_lds_b128)
static __device__ __forceinline__ void cp16_async(const unsigned short* g,
                                                  unsigned short* l) {
  __builtin_amdgcn_global_load_async_to_lds_b128(
      (glob_v4i*)g, (lds_v4i*)l, 0, 0);
}
#else
static __device__ __forceinline__ void cp16_async(const unsigned short* g,
                                                  unsigned short* l) {
  *(uint4*)l = *(const uint4*)g;   // global_load_b128 + ds_store_b128
}
#endif

static __device__ __forceinline__ void wait_async0() {
#if __has_builtin(__builtin_amdgcn_s_wait_asynccnt)
  __builtin_amdgcn_s_wait_asynccnt(0);
#else
  asm volatile("s_wait_asynccnt 0x0" ::: "memory");
#endif
}

// ===========================================================================
// Phase 1: fp32 -> bf16 bulk convert (8 elems/thread)
// ===========================================================================
__global__ __launch_bounds__(256)
void cvt_f32_to_bf16(const float* __restrict__ src,
                     unsigned short* __restrict__ dst, int n8) {
  const int i = blockIdx.x * 256 + threadIdx.x;
  if (i >= n8) return;
  const float4* s4 = (const float4*)src;
  const float4 a = s4[2 * i + 0];
  const float4 b = s4[2 * i + 1];
  uint4 o;
  o.x = f2bf2(a.x, a.y);
  o.y = f2bf2(a.z, a.w);
  o.z = f2bf2(b.x, b.y);
  o.w = f2bf2(b.z, b.w);
  ((uint4*)dst)[i] = o;
}

// ===========================================================================
// Phase 2: bf16 WMMA GEMM + fused LSTM epilogue
//   xb: [4096][2048] bf16,  Wb: [4][2048][2048] bf16 (gate-major: i,f,g,o)
// ===========================================================================
__global__ __launch_bounds__(256)
void lstm_gate_gemm(const unsigned short* __restrict__ xb,
                    const unsigned short* __restrict__ Wb,
                    const float* __restrict__ bi, const float* __restrict__ bf,
                    const float* __restrict__ bg, const float* __restrict__ bo,
                    const float* __restrict__ cx, float* __restrict__ out)
{
  // double-buffered bf16 staging tiles
  __shared__ __align__(16) unsigned short As[2][BM * LDK];       // 2 x 36KB
  __shared__ __align__(16) unsigned short Bs[2][4 * BN * LDK];   // 2 x 18KB

  const int tid  = threadIdx.x;
  const int lane = tid & 31;
  const int wid  = tid >> 5;     // 0..7
  const int mw   = wid >> 1;     // 0..3 : wave M sub-tile (64 rows)
  const int nw   = wid & 1;      // 0..1 : wave N sub-tile (16 cols per gate)
  const int l16  = lane & 15;
  const int hig  = lane >> 4;    // half-wave group, selects K sub-slices

  const int nBase = blockIdx.x * BN;
  const int mBase = blockIdx.y * BM;

  // per-thread staging coordinates (16B chunk per lane per issue)
  const int srow = tid >> 3;            // 0..31 (+32 per issue iter)
  const int scol = (tid & 7) << 3;      // 0,8,..,56

  const v8f vzero = {0.f,0.f,0.f,0.f,0.f,0.f,0.f,0.f};
  v8f acc[4][4];                         // [gate][m-tile]
  #pragma unroll
  for (int g = 0; g < 4; ++g)
    #pragma unroll
    for (int t = 0; t < 4; ++t)
      acc[g][t] = vzero;

  // ---- stage one BK slice into buffer `buf` (12 async b128 per thread) ----
  auto stage = [&](int buf, int k0) {
    #pragma unroll
    for (int it = 0; it < 8; ++it) {                    // x tile: 256 rows
      const int row = srow + it * 32;
      const unsigned short* gp =
          xb + (size_t)(mBase + row) * K_D + k0 + scol;
      cp16_async(gp, &As[buf][row * LDK + scol]);
    }
    #pragma unroll
    for (int it = 0; it < 4; ++it) {                    // W tiles: 4 x 32 rows
      const int row  = srow + it * 32;                  // 0..127 over 4 gates
      const int gate = row >> 5;
      const int wr   = row & 31;
      const unsigned short* gp =
          Wb + ((size_t)gate * N_H + nBase + wr) * K_D + k0 + scol;
      cp16_async(gp, &Bs[buf][row * LDK + scol]);
    }
  };

  stage(0, 0);   // prefetch first slice

  for (int k0 = 0; k0 < K_D; k0 += BK) {
    const int buf = (k0 >> 6) & 1;
    wait_async0();        // my async copies for `buf` have landed
    __syncthreads();      // everyone's copies visible; prev compute retired
    if (k0 + BK < K_D) stage(buf ^ 1, k0 + BK);   // overlap next slice

    #pragma unroll
    for (int kk = 0; kk < BK; kk += 32) {
      // A fragments, 16x32 bf16 (contiguous K, matches B):
      //   lanes 0-15:  K=[kk..kk+15]
      //   lanes 16-31: K=[kk+16..kk+31]
      Frag16 a[4];
      #pragma unroll
      for (int t = 0; t < 4; ++t) {
        const unsigned short* ap =
            &As[buf][(mw * 64 + t * 16 + l16) * LDK + kk + hig * 16];
        a[t].h.lo = *(const v8bf*)(ap);
        a[t].h.hi = *(const v8bf*)(ap + 8);
      }
      #pragma unroll
      for (int g = 0; g < 4; ++g) {
        // B fragment: column n of B == row n of W, 16 contiguous bf16
        Frag16 b;
        const unsigned short* bp =
            &Bs[buf][(g * BN + nw * 16 + l16) * LDK + kk + hig * 16];
        b.h.lo = *(const v8bf*)(bp);
        b.h.hi = *(const v8bf*)(bp + 8);
        #pragma unroll
        for (int t = 0; t < 4; ++t) {
          acc[g][t] = __builtin_amdgcn_wmma_f32_16x16x32_bf16(
              false, a[t].v, false, b.v, (short)0, acc[g][t], false, false);
        }
      }
    }
    __syncthreads();      // reads of `buf` done before it is restaged
  }

  // ---- fused LSTM epilogue ----
  // C/D layout: VGPR r holds row M = r + 8*(lane>=16); N = lane&15.
  const int n = nBase + nw * 16 + l16;
  const float vbi = bi[n], vbf = bf[n], vbg = bg[n], vbo = bo[n];
  const float vcx = cx[n];
  #pragma unroll
  for (int t = 0; t < 4; ++t) {
    #pragma unroll
    for (int r = 0; r < 8; ++r) {
      const int m = mBase + mw * 64 + t * 16 + hig * 8 + r;
      const float gi = sigmoid_f(acc[0][t][r] + vbi);
      const float gf = sigmoid_f(acc[1][t][r] + vbf);
      const float gg = tanh_f   (acc[2][t][r] + vbg);
      const float go = sigmoid_f(acc[3][t][r] + vbo);
      const float c  = gf * vcx + gi * gg;
      out[(size_t)m * N_H + n] = go * tanh_f(c);
    }
  }
}

extern "C" void kernel_launch(void* const* d_in, const int* in_sizes, int n_in,
                              void* d_out, int out_size, void* d_ws, size_t ws_size,
                              hipStream_t stream) {
  const float* x  = (const float*)d_in[0];
  const float* Wi = (const float*)d_in[1];
  const float* bi = (const float*)d_in[2];
  const float* Wf = (const float*)d_in[3];
  const float* bf = (const float*)d_in[4];
  const float* Wg = (const float*)d_in[5];
  const float* bg = (const float*)d_in[6];
  const float* Wo = (const float*)d_in[7];
  const float* bo = (const float*)d_in[8];
  const float* cx = (const float*)d_in[9];

  // workspace layout (bf16): xb [4096*2048], then Wb [4][2048*2048]
  unsigned short* xb = (unsigned short*)d_ws;
  unsigned short* Wb = xb + (size_t)B_M * K_D;
  const size_t wElems = (size_t)N_H * K_D;               // per gate

  // Phase 1: convert once (5 memory-bound launches, same stream)
  {
    const int n8x = (B_M * K_D) / 8;                     // 1,048,576
    cvt_f32_to_bf16<<<dim3((n8x + 255) / 256), dim3(256), 0, stream>>>(x, xb, n8x);
    const int n8w = (int)(wElems / 8);                   // 524,288
    const dim3 gw((n8w + 255) / 256);
    cvt_f32_to_bf16<<<gw, dim3(256), 0, stream>>>(Wi, Wb + 0 * wElems, n8w);
    cvt_f32_to_bf16<<<gw, dim3(256), 0, stream>>>(Wf, Wb + 1 * wElems, n8w);
    cvt_f32_to_bf16<<<gw, dim3(256), 0, stream>>>(Wg, Wb + 2 * wElems, n8w);
    cvt_f32_to_bf16<<<gw, dim3(256), 0, stream>>>(Wo, Wb + 3 * wElems, n8w);
  }

  // Phase 2: fused WMMA GEMM + LSTM cell
  dim3 grid(N_H / BN, B_M / BM);   // (64, 16) = 1024 blocks
  lstm_gate_gemm<<<grid, dim3(256), 0, stream>>>(
      xb, Wb, bi, bf, bg, bo, cx, (float*)d_out);
}
